// PagedSdpaAttention_46471546143195
// MI455X (gfx1250) — matrix operand order repaired and checked by reading, not compiled, not run on hardware
//
#include <hip/hip_runtime.h>
#include <hip/hip_bf16.h>

// ---------------------------------------------------------------------------
// Types
// ---------------------------------------------------------------------------
typedef __bf16 bf16;
typedef __attribute__((ext_vector_type(16))) __bf16 v16bf;
typedef __attribute__((ext_vector_type(8)))  __bf16 v8bf;
typedef __attribute__((ext_vector_type(8)))  float  v8f;

#define S_LEN   2048
#define HID     4096
#define NHEADS  32
#define NKVH    8
#define HDIM    128
#define KVLEN   4096   // 256 blocks * 16
#define NBLK    256    // blocks per request
#define BLKSZ   16

#define KPAD    136    // 128 + 8 bf16 pad: +4 banks per row
#define VPAD    72     // 64 + 8 bf16 pad

static __device__ __forceinline__ v16bf cat8(v8bf lo, v8bf hi) {
    v16bf r;
#pragma unroll
    for (int i = 0; i < 8; ++i) { r[i] = lo[i]; r[i + 8] = hi[i]; }
    return r;
}

static __device__ __forceinline__ v8f wmma_bf16(v16bf a, v16bf b, v8f c) {
    return __builtin_amdgcn_wmma_f32_16x16x32_bf16(
        /*neg_a=*/false, a, /*neg_b=*/false, b,
        /*c_mod=*/(short)0, c, /*reuse_a=*/false, /*reuse_b=*/false);
}

// CDNA5 async global->LDS copy (tracked by ASYNCcnt).
static __device__ __forceinline__ void async_ld_b128(unsigned lds_off,
                                                     const void* gaddr) {
    asm volatile("global_load_async_to_lds_b128 %0, %1, off"
                 :: "v"(lds_off),
                    "v"((unsigned long long)(uintptr_t)gaddr)
                 : "memory");
}
static __device__ __forceinline__ void wait_async0() {
    asm volatile("s_wait_asynccnt 0" ::: "memory");
}

// ---------------------------------------------------------------------------
// fp32 -> bf16 convert (n divisible by 1024)
// ---------------------------------------------------------------------------
__global__ void cvt_f32_bf16(const float* __restrict__ src,
                             bf16* __restrict__ dst, int n) {
    int i = (blockIdx.x * blockDim.x + threadIdx.x) * 4;
    if (i + 3 < n) {
        float4 v = *(const float4*)(src + i);
        dst[i + 0] = (bf16)v.x;
        dst[i + 1] = (bf16)v.y;
        dst[i + 2] = (bf16)v.z;
        dst[i + 3] = (bf16)v.w;
    }
}

// ---------------------------------------------------------------------------
// GEMM: out[M,N] = A[M,K] @ W[N,K]^T   (bf16 in, fp32 accumulate)
// Wave computes a 32x64 tile: 2 A-frags x 4 B-frags -> 8 WMMAs per K-step.
// ---------------------------------------------------------------------------
template <typename OutT>
__global__ __launch_bounds__(256)
void gemm_bf16(const bf16* __restrict__ A, const bf16* __restrict__ W,
               OutT* __restrict__ out, int M, int N, int K) {
    const int lane = threadIdx.x & 31;
    const int wave = threadIdx.x >> 5;
    const int lm = lane & 15;
    const int lh = lane >> 4;
    const int m0 = blockIdx.y * 32;
    const int n0 = blockIdx.x * 512 + wave * 64;

    v8f acc[2][4] = {};
    const bf16* Arow0 = A + (size_t)(m0 + lm) * K;
    const bf16* Arow1 = A + (size_t)(m0 + 16 + lm) * K;

    for (int k = 0; k < K; k += 32) {
        v16bf a0 = cat8(*(const v8bf*)(Arow0 + k + lh * 16),
                        *(const v8bf*)(Arow0 + k + lh * 16 + 8));
        v16bf a1 = cat8(*(const v8bf*)(Arow1 + k + lh * 16),
                        *(const v8bf*)(Arow1 + k + lh * 16 + 8));
#pragma unroll
        for (int j = 0; j < 4; ++j) {
            const bf16* Brow = W + (size_t)(n0 + j * 16 + lm) * K + k + lh * 16;
            v16bf bfrag = *(const v16bf*)Brow;
            acc[0][j] = wmma_bf16(a0, bfrag, acc[0][j]);
            acc[1][j] = wmma_bf16(a1, bfrag, acc[1][j]);
        }
    }
#pragma unroll
    for (int t = 0; t < 2; ++t)
#pragma unroll
        for (int j = 0; j < 4; ++j)
#pragma unroll
            for (int r = 0; r < 8; ++r)
                out[(size_t)(m0 + t * 16 + lh * 8 + r) * N + n0 + j * 16 + lm] =
                    (OutT)acc[t][j][r];
}

// ---------------------------------------------------------------------------
// RoPE in place on Q [S, H*d] bf16, also folds in 1/sqrt(HDIM).
// ---------------------------------------------------------------------------
__global__ void rope_inplace(bf16* __restrict__ Q) {
    int idx = blockIdx.x * blockDim.x + threadIdx.x;   // S*H*64 threads
    int i = idx & 63;
    int h = (idx >> 6) & 31;
    int s = idx >> 11;
    if (s >= S_LEN) return;
    size_t base = (size_t)s * HID + h * HDIM;
    float q0 = (float)Q[base + i];
    float q1 = (float)Q[base + i + 64];
    float ang = (float)s * __expf(-(float)i * (9.210340371976184f / 64.0f));
    float c, sn;
    __sincosf(ang, &sn, &c);
    const float scale = 0.08838834764831845f;  // 1/sqrt(128)
    Q[base + i]      = (bf16)((q0 * c - q1 * sn) * scale);
    Q[base + i + 64] = (bf16)((q1 * c + q0 * sn) * scale);
}

// ---------------------------------------------------------------------------
// Paged KV gather -> Kg [8][4096][128], Vt [8][128][4096] (bf16)
// ---------------------------------------------------------------------------
__global__ __launch_bounds__(256)
void gather_kv(const float* __restrict__ kc, const float* __restrict__ vc,
               const int* __restrict__ btab,
               bf16* __restrict__ Kg, bf16* __restrict__ Vt) {
    int bi  = blockIdx.x >> 3;
    int kvh = blockIdx.x & 7;
    int t = threadIdx.x;
    int r  = t >> 4;          // 0..15 row within block
    int d0 = (t & 15) * 8;    // 8 contiguous d
    int blk = btab[bi];
    size_t src = ((size_t)(blk * NKVH + kvh) * BLKSZ + r) * HDIM + d0;
    size_t kvpos = (size_t)bi * BLKSZ + r;
#pragma unroll
    for (int j = 0; j < 8; ++j) {
        int d = d0 + j;
        Kg[((size_t)kvh * KVLEN + kvpos) * HDIM + d] = (bf16)kc[src + j];
        Vt[(size_t)kvh * HDIM * KVLEN + (size_t)d * KVLEN + kvpos] =
            (bf16)vc[src + j];
    }
}

// ---------------------------------------------------------------------------
// Flash attention. Block = 4 waves = 4 consecutive 16-query tiles of the SAME
// head (4 | 128), so the block stages each 64-kv step's K tile (64x128) and
// V^T tile (128x64) into LDS once via global_load_async_to_lds_b128
// (ASYNCcnt + barrier), then all waves feed WMMA fragments from LDS:
// global/L2 traffic drops 4x (16 -> 64 FLOP/byte).
// Scores transposed: S^T[kv][q] = K_tile @ Q^T; softmax over kv is an in-lane
// reduce + shfl_xor(16) lane-pair exchange; ctx^T += V^T_tile @ P^T.
// 32 WMMAs per 64-kv step per wave.
// ---------------------------------------------------------------------------
__global__ __launch_bounds__(128)
void paged_attn(const bf16* __restrict__ Q,   // [S, HID] (rope'd, pre-scaled)
                const bf16* __restrict__ Kg,  // [NKVH][KVLEN][HDIM]
                const bf16* __restrict__ Vt,  // [NKVH][HDIM][KVLEN]
                bf16* __restrict__ Ctx) {     // [S, HID]
    __shared__ __align__(16) bf16 Ksh[64 * KPAD];    // 17,408 B
    __shared__ __align__(16) bf16 Vsh[HDIM * VPAD];  // 18,432 B

    const int tid  = threadIdx.x;
    const int wid  = blockIdx.x * 4 + (tid >> 5);  // 0..4095
    const int h    = wid >> 7;       // 32 heads (same for all 4 waves)
    const int qt   = wid & 127;      // 128 q-tiles of 16
    const int kvh  = h >> 2;         // GQA factor 4
    const int lane = tid & 31;
    const int lm   = lane & 15;
    const int lh   = lane >> 4;

    // B fragments of Q^T, one per 32-d chunk, reused across all kv.
    v16bf bq[4];
    const bf16* qrow = Q + (size_t)(qt * 16 + lm) * HID + h * HDIM;
#pragma unroll
    for (int c = 0; c < 4; ++c)
        bq[c] = *(const v16bf*)(qrow + c * 32 + lh * 16);

    const bf16* Kh = Kg + (size_t)kvh * KVLEN * HDIM;
    const bf16* Vh = Vt + (size_t)kvh * HDIM * KVLEN;

    float m_run = -1e30f, l_run = 0.0f;
    v8f acc[8] = {};   // ctx^T: 8 d-tiles of 16

    for (int kv = 0; kv < KVLEN; kv += 64) {
        __syncthreads();   // previous tile fully consumed
        // ---- async-stage K (64 rows x 128) and V^T (128 rows x 64) ----
#pragma unroll
        for (int j = 0; j < 8; ++j) {
            int id = tid + j * 128;            // 0..1023
            int kr = id >> 4, kc16 = id & 15;  // K: 16 b128 chunks per row
            async_ld_b128((unsigned)(uintptr_t)&Ksh[kr * KPAD + kc16 * 8],
                          Kh + (size_t)(kv + kr) * HDIM + kc16 * 8);
            int vr = id >> 3, vc8 = id & 7;    // V: 8 b128 chunks per row
            async_ld_b128((unsigned)(uintptr_t)&Vsh[vr * VPAD + vc8 * 8],
                          Vh + (size_t)vr * KVLEN + kv + vc8 * 8);
        }
        wait_async0();
        __syncthreads();   // tile visible to all waves

        // ---- scores^T for four 16-kv tiles (from LDS) ----
        v8f sc[4] = {};
#pragma unroll
        for (int c = 0; c < 4; ++c) {
#pragma unroll
            for (int t = 0; t < 4; ++t) {
                const bf16* kp = &Ksh[(t * 16 + lm) * KPAD + c * 32];
                v16bf ak = cat8(*(const v8bf*)(kp + lh * 16),
                                *(const v8bf*)(kp + lh * 16 + 8));
                sc[t] = wmma_bf16(ak, bq[c], sc[t]);
            }
        }
        // ---- online softmax over kv (column = q = lane&15) ----
        float tmax = sc[0][0];
#pragma unroll
        for (int t = 0; t < 4; ++t)
#pragma unroll
            for (int r = 0; r < 8; ++r) tmax = fmaxf(tmax, sc[t][r]);
        tmax = fmaxf(tmax, __shfl_xor(tmax, 16, 32));
        float mnew  = fmaxf(m_run, tmax);
        float alpha = __expf(m_run - mnew);
        float psum = 0.0f;
        v8f p[4];
#pragma unroll
        for (int t = 0; t < 4; ++t)
#pragma unroll
            for (int r = 0; r < 8; ++r) {
                p[t][r] = __expf(sc[t][r] - mnew);
                psum += p[t][r];
            }
        l_run = l_run * alpha + psum;
        m_run = mnew;
#pragma unroll
        for (int dt = 0; dt < 8; ++dt)
#pragma unroll
            for (int r = 0; r < 8; ++r) acc[dt][r] *= alpha;

        // ---- pack P^T into two 32x16 B fragments (lane-pair exchange) ----
        v16bf bp[2];
        const bool low = (lh == 0);
#pragma unroll
        for (int g = 0; g < 2; ++g) {
            v8f& plo = p[g * 2 + 0];
            v8f& phi = p[g * 2 + 1];
#pragma unroll
            for (int r = 0; r < 8; ++r) {
                float send = low ? phi[r] : plo[r];
                float recv = __shfl_xor(send, 16, 32);
                float e0 = low ? plo[r] : recv;   // elems 0..7
                float e1 = low ? recv : phi[r];   // elems 8..15
                bp[g][r]     = (bf16)e0;
                bp[g][r + 8] = (bf16)e1;
            }
        }
        // ---- ctx^T += V^T @ P^T (from LDS) ----
#pragma unroll
        for (int dt = 0; dt < 8; ++dt) {
            const bf16* vp = &Vsh[(dt * 16 + lm) * VPAD];
            v16bf av0 = cat8(*(const v8bf*)(vp + lh * 16),
                             *(const v8bf*)(vp + lh * 16 + 8));
            v16bf av1 = cat8(*(const v8bf*)(vp + 32 + lh * 16),
                             *(const v8bf*)(vp + 40 + lh * 16));
            acc[dt] = wmma_bf16(av0, bp[0], acc[dt]);
            acc[dt] = wmma_bf16(av1, bp[1], acc[dt]);
        }
    }

    float ltot = l_run + __shfl_xor(l_run, 16, 32);
    float inv = 1.0f / ltot;
    // ctx^T fragment: lane col = q (lm), rows = d (lh*8 + r); store transposed
#pragma unroll
    for (int dt = 0; dt < 8; ++dt)
#pragma unroll
        for (int r = 0; r < 8; ++r)
            Ctx[(size_t)(qt * 16 + lm) * HID + h * HDIM + dt * 16 + lh * 8 + r] =
                (bf16)(acc[dt][r] * inv);
}

// ---------------------------------------------------------------------------
// Launch
// ---------------------------------------------------------------------------
extern "C" void kernel_launch(void* const* d_in, const int* in_sizes, int n_in,
                              void* d_out, int out_size, void* d_ws, size_t ws_size,
                              hipStream_t stream) {
    const float* hidden = (const float*)d_in[0];
    const float* kcache = (const float*)d_in[1];
    const float* vcache = (const float*)d_in[2];
    const float* Wq     = (const float*)d_in[3];
    const float* Wo     = (const float*)d_in[4];
    const int*   btab   = (const int*)d_in[5];
    float* out = (float*)d_out;

    char* w = (char*)d_ws;
    const size_t MB = 1024ull * 1024ull;
    bf16* Xbf = (bf16*)(w);                 // 16 MB  [S,HID]  (reused as Ctx)
    bf16* Wbf = (bf16*)(w + 16 * MB);       // 32 MB  [HID,HID] (Wq then Wo)
    bf16* Qb  = (bf16*)(w + 48 * MB);       // 16 MB  [S,HID]
    bf16* Kg  = (bf16*)(w + 64 * MB);       //  8 MB  [NKVH][KVLEN][HDIM]
    bf16* Vt  = (bf16*)(w + 72 * MB);       //  8 MB  [NKVH][HDIM][KVLEN]
    bf16* Ctx = Xbf;                        // alias: X consumed before attn out

    const int nX = S_LEN * HID;       // 8.4M
    const int nW = HID * HID;         // 16.8M

    // 1) convert activations + Wq to bf16
    cvt_f32_bf16<<<nX / 1024, 256, 0, stream>>>(hidden, Xbf, nX);
    cvt_f32_bf16<<<nW / 1024, 256, 0, stream>>>(Wq, Wbf, nW);

    // 2) Q = X @ Wq^T  (bf16 out)
    dim3 ggrid(HID / 512, S_LEN / 32);
    gemm_bf16<bf16><<<ggrid, 256, 0, stream>>>(Xbf, Wbf, Qb, S_LEN, HID, HID);

    // 3) RoPE (+ softmax scale) in place
    rope_inplace<<<(S_LEN * NHEADS * 64) / 256, 256, 0, stream>>>(Qb);

    // 4) paged KV gather + layout/precision change
    gather_kv<<<NBLK * NKVH, 256, 0, stream>>>(kcache, vcache, btab, Kg, Vt);

    // 5) flash attention -> Ctx bf16 [S,HID]
    paged_attn<<<(NHEADS * (S_LEN / 16)) / 4, 128, 0, stream>>>(Qb, Kg, Vt, Ctx);

    // 6) convert Wo, then out = Ctx @ Wo^T (fp32 out)
    cvt_f32_bf16<<<nW / 1024, 256, 0, stream>>>(Wo, Wbf, nW);
    gemm_bf16<float><<<ggrid, 256, 0, stream>>>(Ctx, Wbf, out, S_LEN, HID, HID);
}
